// MMAttention_6640019439894
// MI455X (gfx1250) — matrix operand order repaired and checked by reading, not compiled, not run on hardware
//
#include <hip/hip_runtime.h>
#include <hip/hip_bf16.h>
#include <math.h>

typedef __attribute__((ext_vector_type(16))) _Float16 v16h;
typedef __attribute__((ext_vector_type(8)))  float    v8f;

#define N_SEQ   16281
#define NPAD    16288              // 16281 padded to 16*1018
#define DIM     512
#define E3      1536
#define HEADS   8
#define DH      64
#define M_PATH  281
#define QKV_SCALE 0.125f           // 64^-0.5
#define PATH_TILES 18              // ceil(281/16)
#define HIST_TILES 1000            // (16281-281)/16 exactly
#define KCHUNK  1024               // split-K chunk (multiple of 32)
#define NCHUNK  16                 // ceil(16288/1024)

// ---------------------------------------------------------------------------
// Kernel 1: QKV GEMM.  qkv[n][e] = sum_d x[n][d] * w[e][d]
// One wave per 16x16 output tile; K=512 -> 16 WMMA steps of 16x16x32 f16.
// Writes q (scaled), k, v head-major [h][NPAD][64] f16 and additionally
// v transposed [h][64][NPAD] so the flash P@V B-fragments are contiguous.
// ---------------------------------------------------------------------------
__global__ void qkv_gemm_kernel(const float* __restrict__ x,
                                const float* __restrict__ w,
                                _Float16* __restrict__ qb,
                                _Float16* __restrict__ kb,
                                _Float16* __restrict__ vb,
                                _Float16* __restrict__ vbt) {
    const int wave = threadIdx.x >> 5;
    const int lane = threadIdx.x & 31;
    const int tile = blockIdx.x * 8 + wave;      // 0 .. 1018*96-1
    const int NT   = E3 / 16;                    // 96
    const int mt   = tile / NT;
    const int nt   = tile % NT;
    const int m    = lane & 15;                  // A row / B col / C col
    const int half = lane >> 4;

    const int row  = mt * 16 + m;
    const int rowc = row < N_SEQ ? row : (N_SEQ - 1);
    const int ecol = nt * 16 + m;

    v8f acc = {};
    #pragma unroll
    for (int k0 = 0; k0 < DIM; k0 += 32) {
        v16h a, b;
        #pragma unroll
        for (int j = 0; j < 16; ++j) {
            const int kk = k0 + half * 16 + j;
            a[j] = (_Float16)x[(size_t)rowc * DIM + kk];
        }
        #pragma unroll
        for (int j = 0; j < 16; ++j) {
            const int kk = k0 + half * 16 + j;
            b[j] = (_Float16)w[(size_t)ecol * DIM + kk];
        }
        acc = __builtin_amdgcn_wmma_f32_16x16x32_f16(false, a, false, b,
                                                     (short)0, acc, false, false);
    }

    // Store: C element c[r] -> row r+8*half, col lane&15
    const int e = nt * 16 + m;
    const int d = e % 64;
    const int h = (e % 512) / 64;
    _Float16* dst;
    float scale = 1.0f;
    bool isV = false;
    if (e < 512)       { dst = qb; scale = QKV_SCALE; }
    else if (e < 1024) { dst = kb; }
    else               { dst = vb; isV = true; }
    #pragma unroll
    for (int r = 0; r < 8; ++r) {
        const int rr = mt * 16 + r + 8 * half;
        const float val = (rr < N_SEQ) ? acc[r] * scale : 0.0f;
        const _Float16 hv = (_Float16)val;
        dst[((size_t)h * NPAD + rr) * DH + d] = hv;
        if (isV) vbt[((size_t)h * DH + d) * NPAD + rr] = hv;
    }
}

// ---------------------------------------------------------------------------
// Kernel 2: flash attention, one wave per (16-query tile x key-chunk),
// 32 keys / iteration.  4 WMMA for scores + 4 WMMA for P@V.  V is read from
// the transposed layout so every lane's B slice is 16 contiguous f16.
// If part_o != nullptr, emits split-K partials (m, l, unnormalized O);
// otherwise normalizes and stores the final rows.
// ---------------------------------------------------------------------------
__global__ void flash_attn_kernel(const _Float16* __restrict__ qb,
                                  const _Float16* __restrict__ kb,
                                  const _Float16* __restrict__ vbt,
                                  float* __restrict__ outb,
                                  float* __restrict__ part_m,
                                  float* __restrict__ part_l,
                                  float* __restrict__ part_o,
                                  int qStart, int qTiles, int qValid,
                                  int nChunks, int chunkSize,
                                  int keyPadTotal, int keyValid) {
    __shared__ _Float16 pl[8][16 * 32];          // per-wave P staging (8 KB)
    const int wave = threadIdx.x >> 5;
    const int lane = threadIdx.x & 31;
    const int g = blockIdx.x * 8 + wave;
    if (g >= HEADS * qTiles * nChunks) return;   // wave-uniform exit
    const int chunk = g % nChunks;
    const int rest  = g / nChunks;
    const int h  = rest / qTiles;
    const int qt = rest % qTiles;
    const int m    = lane & 15;
    const int half = lane >> 4;

    const int keyLo = chunk * chunkSize;
    const int keyHi = (keyLo + chunkSize < keyPadTotal) ? keyLo + chunkSize : keyPadTotal;

    const _Float16* Q  = qb  + (size_t)h * NPAD * DH;
    const _Float16* K  = kb  + (size_t)h * NPAD * DH;
    const _Float16* Vt = vbt + (size_t)h * DH * NPAD;   // [d][key]

    const int qrow = qStart + qt * 16 + m;
    const int qrc  = qrow < NPAD ? qrow : (NPAD - 1);

    // Q A-fragments for d=[0,32) and d=[32,64)
    v16h aq0, aq1;
    #pragma unroll
    for (int j = 0; j < 16; ++j) {
        const int dd = half * 16 + j;
        aq0[j] = Q[(size_t)qrc * DH + dd];
        aq1[j] = Q[(size_t)qrc * DH + 32 + dd];
    }

    float mrow[8], lrow[8];
    v8f o[4] = {};
    #pragma unroll
    for (int r = 0; r < 8; ++r) { mrow[r] = -1e30f; lrow[r] = 0.0f; }

    _Float16* P = &pl[wave][0];

    for (int kb0 = keyLo; kb0 < keyHi; kb0 += 32) {
        // prefetch K four tiles ahead (stream is L2-resident)
        if (kb0 + 128 < keyHi)
            __builtin_prefetch((const void*)(K + (size_t)(kb0 + 128 + m) * DH), 0, 1);

        // ---- score tiles S0 (keys kb0..+15), S1 (keys kb0+16..+31) ----
        v8f s0 = {}, s1 = {};
        {
            v16h b0, b1;
            #pragma unroll
            for (int j = 0; j < 16; ++j) {
                const int dd = half * 16 + j;
                b0[j] = K[(size_t)(kb0 + m) * DH + dd];
                b1[j] = K[(size_t)(kb0 + 16 + m) * DH + dd];
            }
            s0 = __builtin_amdgcn_wmma_f32_16x16x32_f16(false, aq0, false, b0, (short)0, s0, false, false);
            s1 = __builtin_amdgcn_wmma_f32_16x16x32_f16(false, aq0, false, b1, (short)0, s1, false, false);
            #pragma unroll
            for (int j = 0; j < 16; ++j) {
                const int dd = 32 + half * 16 + j;
                b0[j] = K[(size_t)(kb0 + m) * DH + dd];
                b1[j] = K[(size_t)(kb0 + 16 + m) * DH + dd];
            }
            s0 = __builtin_amdgcn_wmma_f32_16x16x32_f16(false, aq1, false, b0, (short)0, s0, false, false);
            s1 = __builtin_amdgcn_wmma_f32_16x16x32_f16(false, aq1, false, b1, (short)0, s1, false, false);
        }

        // ---- mask invalid keys; online softmax update ----
        const bool bad0 = (kb0 + m)      >= keyValid;
        const bool bad1 = (kb0 + 16 + m) >= keyValid;
        #pragma unroll
        for (int r = 0; r < 8; ++r) {
            if (bad0) s0[r] = -1e30f;
            if (bad1) s1[r] = -1e30f;
        }
        #pragma unroll
        for (int r = 0; r < 8; ++r) {
            float v = fmaxf(s0[r], s1[r]);
            v = fmaxf(v, __shfl_xor(v, 1, 32));
            v = fmaxf(v, __shfl_xor(v, 2, 32));
            v = fmaxf(v, __shfl_xor(v, 4, 32));
            v = fmaxf(v, __shfl_xor(v, 8, 32));
            const float mn = fmaxf(mrow[r], v);
            const float alpha = __expf(mrow[r] - mn);
            const float p0 = __expf(s0[r] - mn);
            const float p1 = __expf(s1[r] - mn);
            s0[r] = p0; s1[r] = p1;
            float s = p0 + p1;
            s += __shfl_xor(s, 1, 32);
            s += __shfl_xor(s, 2, 32);
            s += __shfl_xor(s, 4, 32);
            s += __shfl_xor(s, 8, 32);
            lrow[r] = lrow[r] * alpha + s;
            mrow[r] = mn;
            o[0][r] *= alpha; o[1][r] *= alpha; o[2][r] *= alpha; o[3][r] *= alpha;
        }

        // ---- relayout P (C-frag -> A-frag) through per-wave LDS ----
        #pragma unroll
        for (int r = 0; r < 8; ++r) {
            const int row = r + 8 * half;
            P[row * 32 + m]      = (_Float16)s0[r];
            P[row * 32 + 16 + m] = (_Float16)s1[r];
        }
        __builtin_amdgcn_wave_barrier();
        v16h ap;
        #pragma unroll
        for (int j = 0; j < 16; ++j) {
            const int kk = half * 16 + j;
            ap[j] = P[m * 32 + kk];
        }
        __builtin_amdgcn_wave_barrier();

        // ---- O[dtile] += P(16x32) @ V(32x16); contiguous reads from Vt ----
        #pragma unroll
        for (int t = 0; t < 4; ++t) {
            v16h bv;
            #pragma unroll
            for (int j = 0; j < 16; ++j) {
                bv[j] = Vt[(size_t)(t * 16 + m) * NPAD + kb0 + half * 16 + j];
            }
            o[t] = __builtin_amdgcn_wmma_f32_16x16x32_f16(false, ap, false, bv,
                                                          (short)0, o[t], false, false);
        }
    }

    if (part_o) {
        // ---- split-K partial: unnormalized O plus per-row (m, l) ----
        const size_t pi = ((size_t)(h * qTiles + qt) * nChunks + chunk) * 16;
        #pragma unroll
        for (int r = 0; r < 8; ++r) {
            const int rl = r + 8 * half;
            if (m == 0) { part_m[pi + rl] = mrow[r]; part_l[pi + rl] = lrow[r]; }
            float* dst = part_o + (pi + rl) * DH;
            dst[0 * 16 + m] = o[0][r];
            dst[1 * 16 + m] = o[1][r];
            dst[2 * 16 + m] = o[2][r];
            dst[3 * 16 + m] = o[3][r];
        }
    } else {
        // ---- final: normalize and store valid rows ----
        #pragma unroll
        for (int r = 0; r < 8; ++r) {
            const int row = qStart + qt * 16 + r + 8 * half;
            if (row < qValid) {
                const float inv = lrow[r] > 0.0f ? 1.0f / lrow[r] : 0.0f;
                float* dst = outb + ((size_t)h * NPAD + row) * DH;
                dst[0 * 16 + m] = o[0][r] * inv;
                dst[1 * 16 + m] = o[1][r] * inv;
                dst[2 * 16 + m] = o[2][r] * inv;
                dst[3 * 16 + m] = o[3][r] * inv;
            }
        }
    }
}

// ---------------------------------------------------------------------------
// Kernel 3: split-K combine (log-sum-exp merge) for pathway queries.
// One block per (head, query tile).
// ---------------------------------------------------------------------------
__global__ void combine_kernel(const float* __restrict__ part_m,
                               const float* __restrict__ part_l,
                               const float* __restrict__ part_o,
                               float* __restrict__ outb,
                               int qTiles, int nChunks, int qStart, int qValid) {
    __shared__ float sM[16], sL[16];
    const int tile = blockIdx.x;                 // 0 .. HEADS*qTiles-1
    const int h  = tile / qTiles;
    const int qt = tile % qTiles;
    const int tid = threadIdx.x;
    if (tid < 16) {
        float M = -1e30f;
        for (int c = 0; c < nChunks; ++c)
            M = fmaxf(M, part_m[((size_t)tile * nChunks + c) * 16 + tid]);
        float L = 0.0f;
        for (int c = 0; c < nChunks; ++c) {
            const float mc = part_m[((size_t)tile * nChunks + c) * 16 + tid];
            L += part_l[((size_t)tile * nChunks + c) * 16 + tid] * __expf(mc - M);
        }
        sM[tid] = M; sL[tid] = L;
    }
    __syncthreads();
    for (int e = tid; e < 16 * DH; e += 256) {
        const int row = e >> 6, d = e & 63;
        float acc = 0.0f;
        for (int c = 0; c < nChunks; ++c) {
            const float mc = part_m[((size_t)tile * nChunks + c) * 16 + row];
            acc += __expf(mc - sM[row]) *
                   part_o[(((size_t)tile * nChunks + c) * 16 + row) * DH + d];
        }
        const int rowg = qStart + qt * 16 + row;
        if (rowg < qValid) {
            const float inv = sL[row] > 0.0f ? 1.0f / sL[row] : 0.0f;
            outb[((size_t)h * NPAD + rowg) * DH + d] = acc * inv;
        }
    }
}

// ---------------------------------------------------------------------------
// Kernel 4: depthwise conv (k=33) residual on v + transpose to (n, 512) fp32.
// ---------------------------------------------------------------------------
__global__ void conv_out_kernel(const float* __restrict__ attn,
                                const _Float16* __restrict__ vb,
                                const float* __restrict__ wres,
                                float* __restrict__ out) {
    const int idx = blockIdx.x * 256 + threadIdx.x;
    if (idx >= N_SEQ * 512) return;
    const int i = idx / 512;
    const int c = idx % 512;
    const int h = c / DH;
    const int d = c % DH;
    float acc = attn[((size_t)h * NPAD + i) * DH + d];
    #pragma unroll
    for (int t = 0; t < 33; ++t) {
        const int j = i + t - 16;
        if (j >= 0 && j < N_SEQ)
            acc += wres[h * 33 + t] * (float)vb[((size_t)h * NPAD + j) * DH + d];
    }
    out[idx] = acc;
}

// ---------------------------------------------------------------------------
extern "C" void kernel_launch(void* const* d_in, const int* in_sizes, int n_in,
                              void* d_out, int out_size, void* d_ws, size_t ws_size,
                              hipStream_t stream) {
    const float* x    = (const float*)d_in[0];   // (1,16281,512)
    const float* wqkv = (const float*)d_in[1];   // (1536,512)
    const float* wres = (const float*)d_in[2];   // (8,1,33,1) -> 264 floats
    float* out = (float*)d_out;                  // (1,16281,512)

    char* ws = (char*)d_ws;
    const size_t f16Buf = (size_t)HEADS * NPAD * DH * sizeof(_Float16); // ~16.7 MB
    _Float16* qb  = (_Float16*)(ws);
    _Float16* kb  = (_Float16*)(ws + f16Buf);
    _Float16* vb  = (_Float16*)(ws + 2 * f16Buf);
    _Float16* vbt = (_Float16*)(ws + 3 * f16Buf);
    float*   attn = (float*)(ws + 4 * f16Buf);   // [h][NPAD][64] fp32, ~33 MB
    char* p = ws + 4 * f16Buf + (size_t)HEADS * NPAD * DH * sizeof(float);
    const size_t nPartRows = (size_t)HEADS * PATH_TILES * NCHUNK * 16;  // 36864
    float* part_m = (float*)p;                       p += nPartRows * sizeof(float);
    float* part_l = (float*)p;                       p += nPartRows * sizeof(float);
    float* part_o = (float*)p;                       // 36864*64 floats ~ 9.4 MB

    // QKV GEMM: (NPAD/16)*(E3/16) = 1018*96 = 97728 tiles, 8 waves/block
    qkv_gemm_kernel<<<97728 / 8, 256, 0, stream>>>(x, wqkv, qb, kb, vb, vbt);

    // Pathway queries (rows [0,281), softmax over all keys): split-K flash.
    // 8 heads * 18 tiles * 16 chunks = 2304 waves -> 288 blocks.
    flash_attn_kernel<<<(HEADS * PATH_TILES * NCHUNK) / 8, 256, 0, stream>>>(
        qb, kb, vbt, attn, part_m, part_l, part_o,
        /*qStart=*/0, /*qTiles=*/PATH_TILES, /*qValid=*/M_PATH,
        /*nChunks=*/NCHUNK, /*chunkSize=*/KCHUNK,
        /*keyPadTotal=*/NPAD, /*keyValid=*/N_SEQ);

    combine_kernel<<<HEADS * PATH_TILES, 256, 0, stream>>>(
        part_m, part_l, part_o, attn,
        PATH_TILES, NCHUNK, /*qStart=*/0, /*qValid=*/M_PATH);

    // Histology queries (rows [281,16281), keys = 281 pathway tokens):
    // single pass, 8 heads * 1000 tiles = 8000 waves -> 1000 blocks.
    flash_attn_kernel<<<(HEADS * HIST_TILES) / 8, 256, 0, stream>>>(
        qb, kb, vbt, attn, nullptr, nullptr, nullptr,
        /*qStart=*/M_PATH, /*qTiles=*/HIST_TILES, /*qValid=*/N_SEQ,
        /*nChunks=*/1, /*chunkSize=*/288,
        /*keyPadTotal=*/288, /*keyValid=*/M_PATH);

    // Residual conv + transpose-store
    conv_out_kernel<<<(N_SEQ * 512 + 255) / 256, 256, 0, stream>>>(attn, vb, wres, out);
}
